// Logic_Model_55121610277180
// MI455X (gfx1250) — compile-verified, matrix operand order checked by reading, not run on hardware
//
#include <hip/hip_runtime.h>
#include <hip/hip_bf16.h>

typedef __attribute__((ext_vector_type(16))) __bf16    v16bf;
typedef __attribute__((ext_vector_type(8)))  float     v8f;
typedef __attribute__((ext_vector_type(8)))  unsigned  v8ui;
typedef __attribute__((ext_vector_type(4)))  unsigned  v4ui;

constexpr int   L       = 128;
constexpr int   G       = 1667;             // len(arange(0, 50, 0.03))
constexpr float RES_F   = 0.03f;
constexpr float TOL_F   = 0.3f;
constexpr int   NTILE_G = (G + 15) / 16;    // 105 grid tiles
constexpr int   NTILE   = NTILE_G + L / 16; // +8 event tiles = 113

// float -> bf16 bits, round-to-nearest-even (finite inputs only)
__device__ __forceinline__ unsigned short f2bf(float f) {
    unsigned x = __builtin_bit_cast(unsigned, f);
    return (unsigned short)((x + 0x7FFFu + ((x >> 16) & 1u)) >> 16);
}

// #(a[j] < t) over sorted a[0..127], sentinel at a[128]
__device__ __forceinline__ int count_less(const float* a, float t) {
    int lo = 0, hi = 128;
#pragma unroll
    for (int it = 0; it < 8; ++it) { int mid = (lo + hi) >> 1; lo = (a[mid] <  t) ? mid + 1 : lo; hi = (a[mid] <  t) ? hi : mid; }
    return lo;
}
__device__ __forceinline__ int count_leq(const float* a, float t) {
    int lo = 0, hi = 128;
#pragma unroll
    for (int it = 0; it < 8; ++it) { int mid = (lo + hi) >> 1; lo = (a[mid] <= t) ? mid + 1 : lo; hi = (a[mid] <= t) ? hi : mid; }
    return lo;
}

// Keep bf16 halves of packed word u for K = kfirst, kfirst+1 iff K < kcut (prefix mask)
__device__ __forceinline__ unsigned maskpair(unsigned u, int kfirst, int kcut) {
    unsigned r = (kfirst + 1 < kcut) ? u : (u & 0xFFFFu);
    return (kfirst < kcut) ? r : 0u;
}

__global__ __launch_bounds__(256) void loglik_kernel(
    const float* __restrict__ times, const int* __restrict__ states,
    const int* __restrict__ lens, const float* __restrict__ bases,
    const float* __restrict__ weights, float* __restrict__ partial)
{
    __shared__ float  tm0s[129], tm1s[129], tm2s[129];   // sorted times (+sentinel)
    __shared__ float  lastOn0[L], on0f[L];
    __shared__ float  effs1[L], effs2[L];                // +1 if state==0 else -1
    __shared__ __align__(16) unsigned short p1e[L];      // on ? bf16(exp(t)) : 0
    __shared__ __align__(16) unsigned short p2e[L];
    __shared__ float  red[256];

    const int s = blockIdx.x, tid = threadIdx.x;
    const int ln0 = lens[s * 3 + 0], ln1 = lens[s * 3 + 1], ln2 = lens[s * 3 + 2];
    const float b0 = bases[0], b1 = bases[1], b2 = bases[2];
    const float w0 = __expf(weights[0]), w1 = __expf(weights[1]), w2 = __expf(weights[2]);

    if (tid < L) {
        const int j = tid;
        float t0v = times[(s * 3 + 0) * L + j]; tm0s[j] = t0v;
        int   s0  = states[(s * 3 + 0) * L + j];
        on0f[j] = (j < ln0 && s0 == 1) ? 1.f : 0.f;

        float t1v = times[(s * 3 + 1) * L + j]; tm1s[j] = t1v;
        int   s1  = states[(s * 3 + 1) * L + j];
        bool  on1 = (j < ln1 && s1 == 1);
        p1e[j]   = on1 ? f2bf(__expf(t1v)) : (unsigned short)0;
        effs1[j] = (s1 == 0) ? 1.f : -1.f;

        float t2v = times[(s * 3 + 2) * L + j]; tm2s[j] = t2v;
        int   s2  = states[(s * 3 + 2) * L + j];
        bool  on2 = (j < ln2 && s2 == 1);
        p2e[j]   = on2 ? f2bf(__expf(t2v)) : (unsigned short)0;
        effs2[j] = (s2 == 0) ? 1.f : -1.f;
    }
    if (tid == 0) { tm0s[128] = 3e30f; tm1s[128] = 3e30f; tm2s[128] = 3e30f; }
    __syncthreads();
    if (tid == 0) {            // last-on prefix for predicate 0
        float run = -1.0e30f;
        for (int j = 0; j < L; ++j) { if (on0f[j] > 0.5f) run = tm0s[j]; lastOn0[j] = run; }
    }
    __syncthreads();

    float acc = 0.f;

    // ---------- heads 1 & 2: O(log L) per t, per-thread ----------
    auto li12 = [&](float t, float& o1, float& o2) {
        int   k0  = count_leq(tm0s, t);
        float t0r = lastOn0[(k0 > 0) ? (k0 - 1) : 0];     // unconditional load
        float t0l = (k0 > 0) ? t0r : -1.0e30f;
        bool  ex  = t0l > -1.f;
        float d   = t - t0l;
        float ed  = __expf(-d);                           // underflows to 0 when !ex
        float f1  = (ex && (d >  TOL_F)) ? ed : 0.f;      // BEFORE head
        float f2  = (ex && (d <= TOL_F)) ? ed : 0.f;      // EQUAL head
        int k1 = count_less(tm1s, t); int i1 = (k1 > 0) ? (k1 - 1) : (ln1 - 1);
        int k2 = count_less(tm2s, t); int i2 = (k2 > 0) ? (k2 - 1) : (ln2 - 1);
        o1 = b1 + w1 * f1 * effs1[i1];
        o2 = b2 + w2 * f2 * effs2[i2];
    };
    for (int i = tid; i < G; i += 256) {
        float t = i * RES_F, l1, l2;
        li12(t, l1, l2);
        acc -= RES_F * (__expf(l1) + __expf(l2));
    }
    if (tid < L) {
        if (tid >= 1 && tid < ln1) { float l1, l2; li12(tm1s[tid], l1, l2); acc += l1; }
        if (tid >= 1 && tid < ln2) { float l1, l2; li12(tm2s[tid], l1, l2); acc += l2; }
    }

    // ---------- head 0: before-sums as bf16 WMMA row-sum tiles ----------
    const int wave = tid >> 5, lane = tid & 31;
    const int mrow = lane & 15, khalf = lane >> 4;

    v8ui bu;
#pragma unroll
    for (int w = 0; w < 8; ++w) bu[w] = 0x3F803F80u;      // bf16 1.0 pairs
    const v16bf Bones = __builtin_bit_cast(v16bf, bu);

    float acc0 = 0.f;
    for (int tile = wave; tile < NTILE; tile += 8) {      // wave-uniform: EXEC all-ones at WMMA
        const bool isGrid = tile < NTILE_G;
        const int  tbase  = isGrid ? tile * 16 : (tile - NTILE_G) * 16;
        const int   ri    = tbase + mrow;                 // A row = lane&15
        const float tRow  = isGrid ? ri * RES_F : tm0s[ri & 127];
        const float cut   = tRow - TOL_F;
        // Sorted times => eligibility (tm < cut) is a prefix [0, kcut)
        const int kcut1 = count_less(tm1s, cut);
        const int kcut2 = count_less(tm2s, cut);

        v8f D1 = {0.f,0.f,0.f,0.f,0.f,0.f,0.f,0.f};
        v8f D2 = {0.f,0.f,0.f,0.f,0.f,0.f,0.f,0.f};
#pragma unroll
        for (int c = 0; c < 4; ++c) {                     // 4 x 32 events = 128
            // 16-bit A layout: elems 0-7 -> K kb0..kb0+7, elems 8-15 -> K kb1..kb1+7
            const int kb0 = 32 * c + khalf * 8;
            const int kb1 = kb0 + 16;
            v4ui lo1 = *(const v4ui*)&p1e[kb0];
            v4ui hi1 = *(const v4ui*)&p1e[kb1];
            v4ui lo2 = *(const v4ui*)&p2e[kb0];
            v4ui hi2 = *(const v4ui*)&p2e[kb1];
            v8ui a1, a2;
#pragma unroll
            for (int w = 0; w < 4; ++w) {
                a1[w]     = maskpair(lo1[w], kb0 + 2 * w, kcut1);
                a1[w + 4] = maskpair(hi1[w], kb1 + 2 * w, kcut1);
                a2[w]     = maskpair(lo2[w], kb0 + 2 * w, kcut2);
                a2[w + 4] = maskpair(hi2[w], kb1 + 2 * w, kcut2);
            }
            v16bf A1 = __builtin_bit_cast(v16bf, a1);
            v16bf A2 = __builtin_bit_cast(v16bf, a2);
            D1 = __builtin_amdgcn_wmma_f32_16x16x32_bf16(false, A1, false, Bones, (short)0, D1, false, false);
            D2 = __builtin_amdgcn_wmma_f32_16x16x32_bf16(false, A2, false, Bones, (short)0, D2, false, false);
        }
        // D layout: VGPR r = row r (lanes 0-15) / row r+8 (lanes 16-31); rows replicated x16 cols
#pragma unroll
        for (int r = 0; r < 8; ++r) {
            int   rowIdx = tbase + r + 8 * khalf;
            float td = isGrid ? rowIdx * RES_F : tm0s[rowIdx & 127];
            // bs1*bs2 = exp(-2t) * D1 * D2 ; head0 effect == -1 always
            float li = b0 - w0 * __expf(-2.f * td) * D1[r] * D2[r];
            bool  v  = isGrid ? (rowIdx < G) : (rowIdx >= 1 && rowIdx < ln0);
            float contrib = isGrid ? (-RES_F * __expf(li)) : li;
            acc0 += v ? contrib : 0.f;
        }
    }
    acc += acc0 * 0.0625f;   // each row counted by 16 lanes

    // ---------- deterministic block reduction ----------
    red[tid] = acc;
    __syncthreads();
    for (int off = 128; off > 0; off >>= 1) {
        if (tid < off) red[tid] += red[tid + off];
        __syncthreads();
    }
    if (tid == 0) partial[s] = red[0];
}

__global__ __launch_bounds__(256) void reduce_kernel(const float* __restrict__ partial,
                                                     float* __restrict__ out, int n)
{
    __shared__ float red[256];
    float v = 0.f;
    for (int i = threadIdx.x; i < n; i += 256) v += partial[i];
    red[threadIdx.x] = v;
    __syncthreads();
    for (int off = 128; off > 0; off >>= 1) {
        if (threadIdx.x < off) red[threadIdx.x] += red[threadIdx.x + off];
        __syncthreads();
    }
    if (threadIdx.x == 0) out[0] = red[0];
}

extern "C" void kernel_launch(void* const* d_in, const int* in_sizes, int n_in,
                              void* d_out, int out_size, void* d_ws, size_t ws_size,
                              hipStream_t stream)
{
    const float* times   = (const float*)d_in[0];
    const int*   states  = (const int*)d_in[1];
    const int*   lens    = (const int*)d_in[2];
    const float* bases   = (const float*)d_in[3];
    const float* weights = (const float*)d_in[4];
    const int S = in_sizes[2] / 3;            // lens is [S,3]

    float* partial = (float*)d_ws;            // S floats of scratch
    loglik_kernel<<<S, 256, 0, stream>>>(times, states, lens, bases, weights, partial);
    reduce_kernel<<<1, 256, 0, stream>>>(partial, (float*)d_out, S);
}
